// GraphSelfAttention_11484742549822
// MI455X (gfx1250) — matrix operand order repaired and checked by reading, not compiled, not run on hardware
//
#include <hip/hip_runtime.h>
#include <hip/hip_bf16.h>
#include <math.h>

// Problem constants (reference shapes)
#define B_ 4
#define N_ 1024
#define E_ 256
#define H_ 8
#define D_ 32
#define ALPHA 0.1f
#define NEGV -9000000000000000.0f

typedef __attribute__((ext_vector_type(16))) _Float16 v16h;
typedef __attribute__((ext_vector_type(8)))  _Float16 v8h;
typedef __attribute__((ext_vector_type(8)))  float    v8f;

#define DEVI static __device__ __forceinline__

// ---------------- WMMA fragment helpers (wave32, 16x16x32 f16 -> f32) -------

DEVI v16h pack16(float4 f0, float4 f1, float4 f2, float4 f3) {
  v16h a;
  a[0]  = (_Float16)f0.x; a[1]  = (_Float16)f0.y; a[2]  = (_Float16)f0.z; a[3]  = (_Float16)f0.w;
  a[4]  = (_Float16)f1.x; a[5]  = (_Float16)f1.y; a[6]  = (_Float16)f1.z; a[7]  = (_Float16)f1.w;
  a[8]  = (_Float16)f2.x; a[9]  = (_Float16)f2.y; a[10] = (_Float16)f2.z; a[11] = (_Float16)f2.w;
  a[12] = (_Float16)f3.x; a[13] = (_Float16)f3.y; a[14] = (_Float16)f3.z; a[15] = (_Float16)f3.w;
  return a;
}

// A-matrix 16x32 f16: lanes 0-15 -> rows 0-15, K {0..15};
//                     lanes16-31 -> rows 0-15, K {16..31}.
// 16B-aligned vector loads (lda multiple of 32, tiles 16B aligned).
DEVI v16h load_a_f32(const float* A, int lda) {
  int lane = threadIdx.x & 31;
  const float* p = A + (lane & 15) * (size_t)lda + ((lane & 16) ? 16 : 0);
  const float4* p4 = (const float4*)p;
  return pack16(p4[0], p4[1], p4[2], p4[3]);   // p[0..15]
}

// B-matrix 32x16 f16, element B[k][c] stored "col-major": Bm[c*ldb + k]
// (Bm is an (Nout x K) row-major weight; computes X * W^T). 16 contiguous f32.
DEVI v16h load_b_colmaj(const float* Bm, int ldb) {
  int lane = threadIdx.x & 31;
  const float* p = Bm + (lane & 15) * (size_t)ldb + ((lane & 16) ? 16 : 0);
  const float4* p4 = (const float4*)p;
  return pack16(p4[0], p4[1], p4[2], p4[3]);
}

// B-matrix 32x16 f16 from pre-converted f16 "col-major" operand: Bm[c*ldb + k]
// (16 contiguous halves = two 16B loads per lane)
DEVI v16h load_b_f16_colmaj(const _Float16* Bm, int ldb) {
  int lane = threadIdx.x & 31;
  const _Float16* p = Bm + (lane & 15) * (size_t)ldb + ((lane & 16) ? 16 : 0);
  v8h h0 = *(const v8h*)p;
  v8h h1 = *(const v8h*)(p + 8);
  v16h b;
#pragma unroll
  for (int j = 0; j < 8; ++j) { b[j] = h0[j]; b[8 + j] = h1[j]; }
  return b;
}

DEVI v8f wmma16(v16h a, v16h b, v8f c) {
  return __builtin_amdgcn_wmma_f32_16x16x32_f16(
      false, a, false, b, (short)0, c, false, false);
}

// C/D 16x16 f32: VGPR r, lanes0-15 -> M=r, lanes16-31 -> M=8+r, N=lane&15
DEVI void store_c(float* C, int ldc, v8f c) {
  int lane = threadIdx.x & 31;
  int col = lane & 15;
  int rb  = (lane & 16) ? 8 : 0;
#pragma unroll
  for (int r = 0; r < 8; ++r) C[(size_t)(rb + r) * ldc + col] = c[r];
}

// ---------------- tiny converters ------------------------------------------
// WgT[h][d][e] (f16) from Wg[h][e][d]
__global__ void wgt_kernel(const float* __restrict__ Wg, _Float16* __restrict__ WgT) {
  int g = blockIdx.x * blockDim.x + threadIdx.x;   // H*E*D
  int d = g & 31;
  int r = g >> 5;
  int e = r & 255;
  int h = r >> 8;
  WgT[((size_t)h * D_ + d) * E_ + e] = (_Float16)Wg[g];
}

// vT[b][e][n] (f16) from v[b][n][e]
__global__ void vt_kernel(const float* __restrict__ v, _Float16* __restrict__ vT) {
  int g = blockIdx.x * blockDim.x + threadIdx.x;   // B*N*E
  int e = g & 255;
  int r = g >> 8;
  int n = r & 1023;
  int b = r >> 10;
  vT[((size_t)b * E_ + e) * N_ + n] = (_Float16)v[g];
}

// ---------------- Kernel 1: Wh = x @ Wg (f32 out + f16 transposed out) ------
__global__ void wh_kernel(const float* __restrict__ x,
                          const _Float16* __restrict__ WgT,
                          float* __restrict__ Wh,
                          _Float16* __restrict__ WhT) {
  int wid = (int)((blockIdx.x * blockDim.x + threadIdx.x) >> 5);
  int nt = wid & 63;            // N/16 = 64 tiles
  int hb = wid >> 6;            // h*B + b
  int b = hb & 3;
  int h = hb >> 2;
  int n0 = nt * 16;
  const float* Ax = x + ((size_t)b * N_ + n0) * E_;
  const _Float16* Wg0 = WgT + (size_t)h * D_ * E_;         // [d][e]
  v8f c0 = {}; v8f c1 = {};
  for (int k0 = 0; k0 < E_; k0 += 32) {
    v16h a  = load_a_f32(Ax + k0, E_);
    v16h b0 = load_b_f16_colmaj(Wg0 + k0, E_);
    v16h b1 = load_b_f16_colmaj(Wg0 + (size_t)16 * E_ + k0, E_);
    c0 = wmma16(a, b0, c0);
    c1 = wmma16(a, b1, c1);
  }
  float* out = Wh + ((size_t)hb * N_ + n0) * D_;
  store_c(out, D_, c0);
  store_c(out + 16, D_, c1);
  // transposed f16: WhT[hb][d][n]; consecutive r -> consecutive n (one 16B store)
  int lane = threadIdx.x & 31;
  int col = lane & 15;
  int rb  = (lane & 16) ? 8 : 0;
  v8h h0, h1;
#pragma unroll
  for (int r = 0; r < 8; ++r) { h0[r] = (_Float16)c0[r]; h1[r] = (_Float16)c1[r]; }
  *(v8h*)(WhT + ((size_t)hb * D_ + col) * N_ + n0 + rb) = h0;
  *(v8h*)(WhT + ((size_t)hb * D_ + 16 + col) * N_ + n0 + rb) = h1;
}

// ---------------- Kernel 2: s1/s2 = Wh . a1/a2 ------------------------------
__global__ void s12_kernel(const float* __restrict__ Wh,
                           const float* __restrict__ a1,
                           const float* __restrict__ a2,
                           float* __restrict__ s1, float* __restrict__ s2) {
  int i = blockIdx.x * blockDim.x + threadIdx.x;   // H*B*N
  int h = i >> 12;                                 // / (B*N)
  const float4* w  = (const float4*)(Wh + (size_t)i * D_);
  const float4* A1 = (const float4*)(a1 + h * D_);
  const float4* A2 = (const float4*)(a2 + h * D_);
  float u = 0.f, v = 0.f;
#pragma unroll
  for (int d = 0; d < D_ / 4; ++d) {
    float4 t = w[d], p = A1[d], q = A2[d];
    u += t.x * p.x + t.y * p.y + t.z * p.z + t.w * p.w;
    v += t.x * q.x + t.y * q.y + t.z * q.z + t.w * q.w;
  }
  s1[i] = u; s2[i] = v;
}

// ---------------- Kernel 3: fused GAT attention -----------------------------
// block = 256 threads, grid = B * (N/16). dynamic LDS: 16*N floats (64KB)
__global__ void gat_attn_kernel(const float* __restrict__ x,
                                const int*   __restrict__ adj,
                                const _Float16* __restrict__ WhT,
                                const float* __restrict__ s1,
                                const float* __restrict__ s2,
                                float* __restrict__ x1,
                                float* __restrict__ e_out) {
  extern __shared__ float ew[];                    // [16][N_]
  __shared__ float red[16][17];
  __shared__ float rmax[16];
  __shared__ float rinv[16];

  int b  = blockIdx.x >> 6;
  int nt = blockIdx.x & 63;
  int n0 = nt * 16;
  int t = threadIdx.x;
  int row = t >> 4;       // 0..15
  int tc  = t & 15;       // 0..15
  int lane = t & 31;
  int wave = t >> 5;
  int gn = n0 + row;

  // adj mask: head-invariant -> load once, pack 64 bits/thread
  const int* adjr = adj + ((size_t)b * N_ + gn) * N_;
  unsigned long long amask = 0ull;
#pragma unroll 8
  for (int i = 0; i < 64; ++i)
    if (adjr[tc + (i << 4)] > 0) amask |= (1ull << i);

  float acc[64];
#pragma unroll
  for (int i = 0; i < 64; ++i) acc[i] = 0.f;

  for (int h = 0; h < H_; ++h) {
    int hb = h * B_ + b;
    float s1v = s1[(size_t)hb * N_ + gn];
    const float* s2p = s2 + (size_t)hb * N_;

    // pass 1: e = mask(leaky(s1+s2)), row max
    float mx = -3.0e38f;
#pragma unroll 8
    for (int i = 0; i < 64; ++i) {
      int m = tc + (i << 4);
      float e = s1v + s2p[m];
      e = (e >= 0.f) ? e : ALPHA * e;
      e = ((amask >> i) & 1ull) ? e : NEGV;
      ew[row * N_ + m] = e;
      mx = fmaxf(mx, e);
    }
    red[row][tc] = mx;
    __syncthreads();
    if (tc == 0) {
      float m2 = red[row][0];
#pragma unroll
      for (int j = 1; j < 16; ++j) m2 = fmaxf(m2, red[row][j]);
      rmax[row] = m2;
    }
    __syncthreads();
    float m2 = rmax[row];
    float sum = 0.f;
#pragma unroll 8
    for (int i = 0; i < 64; ++i) {
      int m = tc + (i << 4);
      float p = __expf(ew[row * N_ + m] - m2);
      ew[row * N_ + m] = p;
      sum += p;
    }
    red[row][tc] = sum;
    __syncthreads();
    if (tc == 0) {
      float s = 0.f;
#pragma unroll
      for (int j = 0; j < 16; ++j) s += red[row][j];
      rinv[row] = 1.0f / s;
    }
    __syncthreads();
    float is = rinv[row];
#pragma unroll
    for (int i = 0; i < 64; ++i) {
      int m = tc + (i << 4);
      float p = ew[row * N_ + m] * is;
      ew[row * N_ + m] = p;
      acc[i] += p;
    }
    __syncthreads();

    // hp(16x32) = P(16x1024) @ Wh[h,b](1024x32): waves 0,1 via WMMA
    // A from LDS (f32->f16), B from WhT (f16, two 16B loads per fragment)
    if (wave < 2) {
      int dcol = wave * 16;
      const _Float16* WT = WhT + ((size_t)hb * D_ + dcol) * N_;
      v8f c = {};
      for (int k0 = 0; k0 < N_; k0 += 32) {
        v16h a  = load_a_f32(ew + k0, N_);
        v16h bf = load_b_f16_colmaj(WT + k0, N_);
        c = wmma16(a, bf, c);
      }
      int col = lane & 15;
      int rb  = (lane & 16) ? 8 : 0;
#pragma unroll
      for (int r = 0; r < 8; ++r) {
        float v = c[r];
        v = (v >= 0.f) ? v : 0.01f * v;
        size_t off = ((size_t)b * N_ + (n0 + rb + r)) * E_ + h * D_ + dcol + col;
        x1[off] = x[off] + v;
      }
    }
    __syncthreads();
  }
  // e_out[b, gn, :] = mean over heads
  float* ep = e_out + ((size_t)b * N_ + gn) * N_;
#pragma unroll 8
  for (int i = 0; i < 64; ++i) ep[tc + (i << 4)] = acc[i] * (1.0f / (float)H_);
}

// ---------------- cnorm over features (axis=2), in place --------------------
__global__ void cnorm_feat_kernel(float* __restrict__ xb) {
  int wid = (int)((blockIdx.x * blockDim.x + threadIdx.x) >> 5);  // row id
  int lane = threadIdx.x & 31;
  float* p = xb + (size_t)wid * E_ + lane * 8;     // 32B contiguous per lane
  float4 a = ((const float4*)p)[0];
  float4 bq = ((const float4*)p)[1];
  float s = a.x + a.y + a.z + a.w + bq.x + bq.y + bq.z + bq.w;
#pragma unroll
  for (int o = 16; o > 0; o >>= 1) s += __shfl_xor(s, o);
  float mean = s * (1.0f / E_);
  float v[8] = { a.x, a.y, a.z, a.w, bq.x, bq.y, bq.z, bq.w };
  float var = 0.f;
#pragma unroll
  for (int i = 0; i < 8; ++i) { float d = v[i] - mean; var += d * d; }
#pragma unroll
  for (int o = 16; o > 0; o >>= 1) var += __shfl_xor(var, o);
  float inv = 1.0f / (sqrtf(var * (1.0f / (E_ - 1))) + 1e-6f);
  float4 o0, o1;
  o0.x = (v[0] - mean) * inv; o0.y = (v[1] - mean) * inv;
  o0.z = (v[2] - mean) * inv; o0.w = (v[3] - mean) * inv;
  o1.x = (v[4] - mean) * inv; o1.y = (v[5] - mean) * inv;
  o1.z = (v[6] - mean) * inv; o1.w = (v[7] - mean) * inv;
  ((float4*)p)[0] = o0;
  ((float4*)p)[1] = o1;
}

// ---------------- cnorm over rows (axis=1), in place ------------------------
__global__ void cnorm_col_kernel(float* __restrict__ xb) {
  int g = blockIdx.x * blockDim.x + threadIdx.x;   // B*E threads
  int b = g >> 8;
  int e = g & 255;
  float* p = xb + (size_t)b * N_ * E_ + e;
  float s = 0.f, ss = 0.f;
  for (int n = 0; n < N_; ++n) { float t = p[(size_t)n * E_]; s += t; ss += t * t; }
  float mean = s * (1.0f / N_);
  float var = (ss - s * mean) * (1.0f / (N_ - 1));
  var = fmaxf(var, 0.f);
  float inv = 1.0f / (sqrtf(var) + 1e-6f);
  for (int n = 0; n < N_; ++n) {
    size_t o = (size_t)n * E_;
    p[o] = (p[o] - mean) * inv;
  }
}

// ---------------- generic GEMM: Out = A(Mtot x K) @ W(Nout x K)^T + bias ----
__global__ void gemm_bias_kernel(const float* __restrict__ A,
                                 const float* __restrict__ W,
                                 const float* __restrict__ bias,
                                 float* __restrict__ Out,
                                 int Mtot, int Nout, int K) {
  int wid = (int)((blockIdx.x * blockDim.x + threadIdx.x) >> 5);
  int ntiles = Nout >> 4;
  int mt = wid / ntiles;
  int nt = wid % ntiles;
  const float* Ab = A + (size_t)(mt * 16) * K;
  const float* Wb = W + (size_t)(nt * 16) * K;
  v8f c = {};
  for (int k0 = 0; k0 < K; k0 += 32) {
    v16h a  = load_a_f32(Ab + k0, K);
    v16h bf = load_b_colmaj(Wb + k0, K);
    c = wmma16(a, bf, c);
  }
  int lane = threadIdx.x & 31;
  int col = lane & 15;
  int rb  = (lane & 16) ? 8 : 0;
  float bv = bias ? bias[nt * 16 + col] : 0.f;
  float* o = Out + (size_t)(mt * 16) * Nout + nt * 16;
#pragma unroll
  for (int r = 0; r < 8; ++r) o[(size_t)(rb + r) * Nout + col] = c[r] + bv;
}

// ---------------- src (N,B,E) -> srcb (B,N,E) -------------------------------
__global__ void srcb_kernel(const float* __restrict__ src, float* __restrict__ srcb) {
  int g = blockIdx.x * blockDim.x + threadIdx.x;   // B*N*E
  int e = g & (E_ - 1);
  int r = g >> 8;
  int n = r & (N_ - 1);
  int b = r >> 10;
  srcb[g] = src[((size_t)n * B_ + b) * E_ + e];
}

// ---------------- MHA: per (b,h,qtile16) fused scores+softmax+PV ------------
__global__ void mha_attn_kernel(const float* __restrict__ q,
                                const float* __restrict__ k,
                                const _Float16* __restrict__ vT,
                                float* __restrict__ o) {
  extern __shared__ float sm[];                    // [16][N_]
  __shared__ float red[16][17];
  __shared__ float rmax[16];
  __shared__ float rinv[16];

  int qt = blockIdx.x & 63;
  int bh = blockIdx.x >> 6;
  int h = bh & 7;
  int b = bh >> 3;
  int n0 = qt * 16;
  int t = threadIdx.x;
  int lane = t & 31;
  int wave = t >> 5;
  const float scale = 0.1767766952966369f;  // 1/sqrt(32)

  // S = q_tile(16x32) @ k^T(32xN): one WMMA per 16-col chunk (K=32)
  const float* qb = q + ((size_t)b * N_ + n0) * E_ + h * 32;
  v16h a_q = load_a_f32(qb, E_);
  for (int mc = wave; mc < 64; mc += 8) {
    const float* kb = k + ((size_t)b * N_ + mc * 16) * E_ + h * 32;
    v16h bf = load_b_colmaj(kb, E_);
    v8f c = {};
    c = wmma16(a_q, bf, c);
    int col = lane & 15;
    int rb  = (lane & 16) ? 8 : 0;
#pragma unroll
    for (int r = 0; r < 8; ++r) sm[(rb + r) * N_ + mc * 16 + col] = c[r] * scale;
  }
  __syncthreads();

  // softmax rows
  int row = t >> 4;
  int tc  = t & 15;
  float mx = -3.0e38f;
#pragma unroll 8
  for (int i = 0; i < 64; ++i) mx = fmaxf(mx, sm[row * N_ + tc + (i << 4)]);
  red[row][tc] = mx;
  __syncthreads();
  if (tc == 0) {
    float m2 = red[row][0];
#pragma unroll
    for (int j = 1; j < 16; ++j) m2 = fmaxf(m2, red[row][j]);
    rmax[row] = m2;
  }
  __syncthreads();
  float m2 = rmax[row];
  float sum = 0.f;
#pragma unroll 8
  for (int i = 0; i < 64; ++i) {
    int m = tc + (i << 4);
    float p = __expf(sm[row * N_ + m] - m2);
    sm[row * N_ + m] = p;
    sum += p;
  }
  red[row][tc] = sum;
  __syncthreads();
  if (tc == 0) {
    float s = 0.f;
#pragma unroll
    for (int j = 0; j < 16; ++j) s += red[row][j];
    rinv[row] = 1.0f / s;
  }
  __syncthreads();
  float is = rinv[row];
#pragma unroll 8
  for (int i = 0; i < 64; ++i) sm[row * N_ + tc + (i << 4)] *= is;
  __syncthreads();

  // O(16x32) = P(16xN) @ V(Nx32): waves 0,1; B from vT (f16 contiguous)
  if (wave < 2) {
    int dcol = wave * 16;
    const _Float16* vbT = vT + ((size_t)b * E_ + h * 32 + dcol) * N_;
    v8f c = {};
    for (int k0 = 0; k0 < N_; k0 += 32) {
      v16h a  = load_a_f32(sm + k0, N_);
      v16h bf = load_b_f16_colmaj(vbT + k0, N_);
      c = wmma16(a, bf, c);
    }
    int col = lane & 15;
    int rb  = (lane & 16) ? 8 : 0;
#pragma unroll
    for (int r = 0; r < 8; ++r)
      o[((size_t)b * N_ + n0 + rb + r) * E_ + h * 32 + dcol + col] = c[r];
  }
}

// ---------------- host-side orchestration -----------------------------------
extern "C" void kernel_launch(void* const* d_in, const int* in_sizes, int n_in,
                              void* d_out, int out_size, void* d_ws, size_t ws_size,
                              hipStream_t stream) {
  const float* x          = (const float*)d_in[0];
  const float* src        = (const float*)d_in[1];
  const int*   adj        = (const int*)  d_in[2];
  const float* Wg         = (const float*)d_in[3];
  const float* a1         = (const float*)d_in[4];
  const float* a2         = (const float*)d_in[5];
  const float* lin_w      = (const float*)d_in[6];
  const float* lin_b      = (const float*)d_in[7];
  const float* in_proj_w  = (const float*)d_in[8];
  const float* in_proj_b  = (const float*)d_in[9];
  const float* out_proj_w = (const float*)d_in[10];
  const float* out_proj_b = (const float*)d_in[11];

  float* out_x = (float*)d_out;                                // B*N*E
  float* e_out = (float*)d_out + (size_t)B_ * N_ * E_;         // B*N*N

  const size_t M1 = (size_t)H_ * B_ * N_ * D_;                 // 1M
  const size_t MX = (size_t)B_ * N_ * E_;                      // 1M
  float* ws   = (float*)d_ws;
  float* Wh   = ws;                            // M1 f32
  float* s1   = Wh + M1;                       // 32K
  float* s2   = s1 + (size_t)H_ * B_ * N_;     // 32K
  float* x1   = s2 + (size_t)H_ * B_ * N_;     // MX
  float* x2   = x1 + MX;                       // MX (x2 -> x3 in place)
  float* srcb = x2 + MX;                       // MX
  float* qb   = srcb + MX;                     // MX
  float* kb   = qb + MX;                       // MX
  float* vb   = kb + MX;                       // MX
  float* ob   = vb + MX;                       // MX
  _Float16* WhT = (_Float16*)(ob + MX);        // M1 halves (0.5M floats)
  _Float16* vT  = (_Float16*)((float*)WhT + M1 / 2);   // MX halves
  _Float16* WgT = (_Float16*)((float*)vT + MX / 2);    // H*E*D halves

  const size_t smem = 16 * N_ * sizeof(float);  // 64KB

  // 0) weight transpose/convert
  wgt_kernel<<<(H_ * E_ * D_) / 256, 256, 0, stream>>>(Wg, WgT);
  // 1) Wh = x @ Wg  (H*B*64 waves) -> Wh f32 + WhT f16
  wh_kernel<<<256, 256, 0, stream>>>(x, WgT, Wh, WhT);
  // 2) s1, s2
  s12_kernel<<<(H_ * B_ * N_) / 256, 256, 0, stream>>>(Wh, a1, a2, s1, s2);
  // 3) fused GAT attention -> x1, e_out
  gat_attn_kernel<<<B_ * (N_ / 16), 256, smem, stream>>>(x, adj, WhT, s1, s2, x1, e_out);
  // 4) norm(x1) in place (axis=2 then axis=1)
  cnorm_feat_kernel<<<(B_ * N_) / 8, 256, 0, stream>>>(x1);
  cnorm_col_kernel<<<(B_ * E_) / 256, 256, 0, stream>>>(x1);
  // 5) x2 = x1 @ lin_w^T + lin_b
  {
    int waves = ((B_ * N_) / 16) * (E_ / 16);
    gemm_bias_kernel<<<waves / 8, 256, 0, stream>>>(x1, lin_w, lin_b, x2,
                                                    B_ * N_, E_, E_);
  }
  // 6) norm(x2) in place
  cnorm_feat_kernel<<<(B_ * N_) / 8, 256, 0, stream>>>(x2);
  cnorm_col_kernel<<<(B_ * E_) / 256, 256, 0, stream>>>(x2);
  // 7) srcb = transpose(src)
  srcb_kernel<<<(int)(MX / 256), 256, 0, stream>>>(src, srcb);
  // 8) q/k/v projections
  {
    int waves = ((B_ * N_) / 16) * (E_ / 16);
    gemm_bias_kernel<<<waves / 8, 256, 0, stream>>>(x2, in_proj_w,
                                                    in_proj_b, qb, B_ * N_, E_, E_);
    gemm_bias_kernel<<<waves / 8, 256, 0, stream>>>(x2, in_proj_w + (size_t)E_ * E_,
                                                    in_proj_b + E_, kb, B_ * N_, E_, E_);
    gemm_bias_kernel<<<waves / 8, 256, 0, stream>>>(srcb, in_proj_w + 2 * (size_t)E_ * E_,
                                                    in_proj_b + 2 * E_, vb, B_ * N_, E_, E_);
  }
  // 8b) vT = f16 transpose of v (contiguous-K B operand for P@V)
  vt_kernel<<<(int)(MX / 256), 256, 0, stream>>>(vb, vT);
  // 9) fused MHA -> ob
  mha_attn_kernel<<<B_ * H_ * (N_ / 16), 256, smem, stream>>>(qb, kb, vT, ob);
  // 10) out = ob @ out_proj_w^T + out_proj_b
  {
    int waves = ((B_ * N_) / 16) * (E_ / 16);
    gemm_bias_kernel<<<waves / 8, 256, 0, stream>>>(ob, out_proj_w, out_proj_b,
                                                    out_x, B_ * N_, E_, E_);
  }
}
